// BilinearConvLinear_34213709480488
// MI455X (gfx1250) — matrix operand order repaired and checked by reading, not compiled, not run on hardware
//
#include <hip/hip_runtime.h>
#include <hip/hip_bf16.h>

typedef __attribute__((ext_vector_type(16))) _Float16 v16h;
typedef __attribute__((ext_vector_type(8)))  _Float16 v8h;
typedef __attribute__((ext_vector_type(4)))  _Float16 v4h;
typedef __attribute__((ext_vector_type(8)))  float    v8f;

#define BB 4
#define TT 4096
#define CC 64
#define LL 15
#define PADW 7
#define NN 64
#define BT (BB*TT)          // 16384
#define EPSV 1e-12f

// ---------------------------------------------------------------------------
// Kernel 1: stage conv_w (4096 x 64 f32) into f16 WMMA B-fragments.
// B tile for (ntile nt, ktile kt): 32x16 f16.  Per-lane layout (wave32):
//   lane<16  -> column n = nt*16 + lane,      16 halves = K  0..15 (of kt*32)
//   lane>=16 -> column n = nt*16 + (lane-16), 16 halves = K 16..31
// Stored so the GEMM does ONE contiguous 32B load per fragment per lane.
// ---------------------------------------------------------------------------
__global__ void stage_b_kernel(const float* __restrict__ conv_w,
                               _Float16* __restrict__ bstage) {
    int idx  = blockIdx.x * 256 + threadIdx.x;   // 0 .. 262143
    int h    = idx & 15;
    int lane = (idx >> 4) & 31;
    int kt   = (idx >> 9) & 127;
    int nt   = idx >> 16;
    int n    = nt * 16 + (lane & 15);
    int K    = kt * 32 + (lane >> 4) * 16 + h;
    bstage[idx] = (_Float16)conv_w[K * NN + n];
}

// ---------------------------------------------------------------------------
// Kernel 2: windowed dots d[b,s,j] = x[b,s,:] . x[b,s+j,:]  (0 if s+j >= T)
// ---------------------------------------------------------------------------
__global__ void dot_kernel(const float* __restrict__ x, float* __restrict__ dws) {
    int idx = blockIdx.x * 256 + threadIdx.x;    // 0 .. 245759
    int j   = idx % LL;
    int bs  = idx / LL;                          // b*T + s
    int s   = bs & (TT - 1);
    float acc = 0.f;
    if (s + j < TT) {
        const float4* p0 = (const float4*)(x + (size_t)bs * CC);
        const float4* p1 = (const float4*)(x + (size_t)(bs + j) * CC);
        #pragma unroll
        for (int i = 0; i < 16; ++i) {
            float4 a = p0[i], b = p1[i];
            acc += a.x * b.x + a.y * b.y + a.z * b.z + a.w * b.w;
        }
    }
    dws[idx] = acc;
}

// ---------------------------------------------------------------------------
// Kernel 3: sq[b,t] = sum_{l,m} w_l w_m (x_{t+l} . x_{t+m})^2  (Gram trick)
// ---------------------------------------------------------------------------
__global__ void sq_kernel(const float* __restrict__ dws,
                          const float* __restrict__ w,
                          float* __restrict__ sqws) {
    int idx = blockIdx.x * 256 + threadIdx.x;    // b*T + t, 16384 total
    int t    = idx & (TT - 1);
    int base = idx & ~(TT - 1);
    float wl[LL];
    #pragma unroll
    for (int l = 0; l < LL; ++l) wl[l] = w[l];
    float acc = 0.f;
    #pragma unroll
    for (int l = 0; l < LL; ++l) {
        int s1 = t + l - PADW;
        if (s1 < 0 || s1 >= TT) continue;        // zero-padded window
        const float* drow = dws + (size_t)(base + s1) * LL;
        float d0 = drow[0];
        acc += wl[l] * wl[l] * d0 * d0;
        for (int j = 1; j <= 14 - l; ++j) {      // d already 0 past T
            float dj = drow[j];
            acc += 2.f * wl[l] * wl[l + j] * dj * dj;
        }
    }
    sqws[idx] = acc;
}

// ---------------------------------------------------------------------------
// Kernel 4: q[s,n] = F[s,:] @ conv_w[:,n],  F[s, c*64+d] = x[s,c]*x[s,d]
//
// Block = 256 threads (8 waves), M = 64 timesteps, N = 64.
//   wave w: N-tile nt = w & 3, M-pair mtb = (w>>2)*2  -> two 16-row M-tiles
//   sharing one B fragment in registers (2 WMMAs per B load: halves L2
//   traffic for conv_w and doubles wmma density vs. round-1).
// K = 4096 in 8 chunks of 512 (c-groups of 8), 16 WMMA k-steps per chunk.
// Rank-1 feature chunk built in LDS with packed f16 math; never hits HBM.
// LDS: 8 KB x-tile (f16) + 64 KB F-chunk = 72 KB.
// ---------------------------------------------------------------------------
__global__ __launch_bounds__(256) void qgemm_kernel(
        const float* __restrict__ x,
        const v16h*  __restrict__ bfrag,
        float*       __restrict__ q) {
    __shared__ __attribute__((aligned(16))) _Float16 xh[64][CC];    //  8 KB
    __shared__ __attribute__((aligned(16))) _Float16 Fsh[64][512];  // 64 KB

    const int tid  = threadIdx.x;
    const int wave = tid >> 5;
    const int lane = tid & 31;
    const int nt   = wave & 3;          // N-tile 0..3
    const int mtb  = (wave >> 2) * 2;   // first of two M-tiles: 0 or 2
    const int m    = lane & 15;
    const int half = lane >> 4;
    const int r0   = blockIdx.x * 64;   // global row base (b*T+s)

    // load 64x64 f32 x tile, convert to f16 (1024 float4 -> 1024 v4h)
    for (int i = tid; i < 1024; i += 256) {
        float4 v = ((const float4*)(x + (size_t)r0 * CC))[i];
        v4h h;
        h[0] = (_Float16)v.x; h[1] = (_Float16)v.y;
        h[2] = (_Float16)v.z; h[3] = (_Float16)v.w;
        ((v4h*)&xh[0][0])[i] = h;
    }
    __syncthreads();

    v8f acc0 = {}, acc1 = {};

    for (int cg = 0; cg < 8; ++cg) {              // c-group: c in [cg*8, cg*8+8)
        // build F chunk: Fsh[mr][(c-cg*8)*64 + d] = xh[mr][c] * xh[mr][d]
        // packed f16 products (v_pk_mul_f16), 512 combos / 256 threads
        for (int combo = tid; combo < 512; combo += 256) {
            const int mr = combo >> 3;
            const int cl = combo & 7;
            const _Float16 xc = xh[mr][cg * 8 + cl];
            const v4h xc4 = {xc, xc, xc, xc};
            const v4h* xrow = (const v4h*)&xh[mr][0];
            v4h* row = (v4h*)&Fsh[mr][cl * 64];
            #pragma unroll
            for (int d4 = 0; d4 < 16; ++d4)
                row[d4] = xc4 * xrow[d4];
        }
        __syncthreads();

        // 16 WMMA k-steps over this 512-wide chunk; B fragment reused by
        // both M-tiles of this wave.
        const v16h* bp = bfrag + (size_t)(nt * 128 + cg * 16) * 32 + lane;
        #pragma unroll 4
        for (int ks = 0; ks < 16; ++ks) {
            v16h bf = bp[(size_t)ks * 32];
            // A fragments in contiguous k order, same as B staging:
            // lane<16: halves = [K0..15]; lane>=16: [K16..31]
            const v8h* r0p = (const v8h*)&Fsh[mtb * 16 + m][ks * 32];
            const v8h* r1p = (const v8h*)&Fsh[mtb * 16 + 16 + m][ks * 32];
            v8h lo0 = r0p[2 * half], hi0 = r0p[2 * half + 1];
            v8h lo1 = r1p[2 * half], hi1 = r1p[2 * half + 1];
            v16h a0, a1;
            #pragma unroll
            for (int i = 0; i < 8; ++i) {
                a0[i] = lo0[i]; a0[8 + i] = hi0[i];
                a1[i] = lo1[i]; a1[8 + i] = hi1[i];
            }
            acc0 = __builtin_amdgcn_wmma_f32_16x16x32_f16(
                       false, a0, false, bf, (short)0, acc0, false, false);
            acc1 = __builtin_amdgcn_wmma_f32_16x16x32_f16(
                       false, a1, false, bf, (short)0, acc1, false, false);
        }
        __syncthreads();
    }

    // C/D layout: VGPR j -> row m = j + half*8, col n = nt*16 + (lane&15)
    const int n = nt * 16 + m;
    const int rb0 = r0 + mtb * 16 + half * 8;
    const int rb1 = rb0 + 16;
    #pragma unroll
    for (int j = 0; j < 8; ++j) {
        q[(size_t)(rb0 + j) * NN + n] = acc0[j];
        q[(size_t)(rb1 + j) * NN + n] = acc1[j];
    }
}

// ---------------------------------------------------------------------------
// Kernel 5: out[b,t,n] = rsqrt(max(sq,eps)) * sum_l w_l q[b,t+l-7,n] + bias
// ---------------------------------------------------------------------------
__global__ void out_kernel(const float* __restrict__ q,
                           const float* __restrict__ sqws,
                           const float* __restrict__ w,
                           const float* __restrict__ conv_b,
                           float* __restrict__ out) {
    int tid = threadIdx.x;             // 256
    int n   = tid & 63;
    int idx = blockIdx.x * 4 + (tid >> 6);   // b*T + t
    int t    = idx & (TT - 1);
    int base = idx & ~(TT - 1);
    float acc = 0.f;
    #pragma unroll
    for (int l = 0; l < LL; ++l) {
        int s1 = t + l - PADW;
        if (s1 >= 0 && s1 < TT)
            acc += w[l] * q[(size_t)(base + s1) * NN + n];
    }
    float r = __frsqrt_rn(fmaxf(sqws[idx], EPSV));
    out[(size_t)idx * NN + n] = acc * r + conv_b[n];
}

// ---------------------------------------------------------------------------
extern "C" void kernel_launch(void* const* d_in, const int* in_sizes, int n_in,
                              void* d_out, int out_size, void* d_ws, size_t ws_size,
                              hipStream_t stream) {
    const float* x      = (const float*)d_in[0];   // B*T*C
    const float* w      = (const float*)d_in[1];   // 15
    const float* conv_w = (const float*)d_in[2];   // 4096*64
    const float* conv_b = (const float*)d_in[3];   // 64
    float* out = (float*)d_out;                    // B*T*64

    // workspace carve (~5.7 MB total)
    char* p = (char*)d_ws;
    _Float16* bstage = (_Float16*)p;               p += 4 * 128 * 32 * 16 * sizeof(_Float16); // 512 KB
    float*    qws    = (float*)p;                  p += (size_t)BT * NN * sizeof(float);      // 4 MB
    float*    dws    = (float*)p;                  p += (size_t)BT * LL * sizeof(float);      // 983 KB
    float*    sqws   = (float*)p;                                                             // 64 KB

    stage_b_kernel<<<1024, 256, 0, stream>>>(conv_w, bstage);
    dot_kernel    <<<(BT * LL) / 256, 256, 0, stream>>>(x, dws);
    sq_kernel     <<<BT / 256, 256, 0, stream>>>(dws, w, sqws);
    qgemm_kernel  <<<BT / 64, 256, 0, stream>>>(x, (const v16h*)bstage, qws);
    out_kernel    <<<BT / 4, 256, 0, stream>>>(qws, sqws, w, conv_b, out);
}
